// Decoder_61108794688068
// MI455X (gfx1250) — matrix operand order repaired and checked by reading, not compiled, not run on hardware
//
#include <hip/hip_runtime.h>
#include <math.h>

typedef __attribute__((ext_vector_type(16))) _Float16 v16h;
typedef __attribute__((ext_vector_type(8)))  _Float16 v8h;
typedef __attribute__((ext_vector_type(8)))  float    v8f;

#define NPTS 4096
#define EPSN 1e-5f
#define BM 64
#define BNT 64
#define BK 32

// ---------------------------------------------------------------------------
// BN batch-stats folded to scale/shift: y = x*scale[c] + shift[c]
// ---------------------------------------------------------------------------
__global__ __launch_bounds__(256) void bn_stats_kernel(
    const float* __restrict__ X, int B, int C, int N,
    const float* __restrict__ g, const float* __restrict__ b,
    float* __restrict__ scale, float* __restrict__ shift)
{
  int c = blockIdx.x;
  float s = 0.f, s2 = 0.f;
  for (int bb = 0; bb < B; ++bb) {
    const float* Xp = X + ((size_t)bb * C + c) * N;
    for (int n = threadIdx.x; n < N; n += 256) {
      float v = Xp[n];
      s += v; s2 += v * v;
    }
  }
  __shared__ float rs[256], rq[256];
  rs[threadIdx.x] = s; rq[threadIdx.x] = s2;
  __syncthreads();
  for (int off = 128; off > 0; off >>= 1) {
    if ((int)threadIdx.x < off) {
      rs[threadIdx.x] += rs[threadIdx.x + off];
      rq[threadIdx.x] += rq[threadIdx.x + off];
    }
    __syncthreads();
  }
  if (threadIdx.x == 0) {
    float total = (float)B * (float)N;
    float m   = rs[0] / total;
    float var = rq[0] / total - m * m;
    float sc  = g[c] * rsqrtf(var + EPSN);
    scale[c] = sc;
    shift[c] = b[c] - m * sc;
  }
}

// ---------------------------------------------------------------------------
// WMMA GEMM: C = A(MxK) * B(KxN) [+ bias[m]] [+ addMat[m,n]]
//  BMODE 0: B raw f32 (KxN row-major).  Requires N%64==0 and ldb%4==0.
//  BMODE 1: B = relu(raw*bnScale[k]+bnShift[k]).  Same requirements.
//  BMODE 2: B given as (NxK) row-major => A * B2^T.
// Branch-free clamped staging (no exec divergence), f16 LDS tiles, f32 acc.
// Block 64x64, 8 waves; each wave: 1 A-frag reused over 2 WMMAs per K-step.
// ---------------------------------------------------------------------------
template<int BMODE, bool HASBIAS, bool HASADD>
__global__ __launch_bounds__(256) void gemm_wmma(
    const float* __restrict__ A, long long aStride, int lda,
    const float* __restrict__ B, long long bStride, int ldb,
    float* __restrict__ C, long long cStride, int ldc,
    const float* __restrict__ bias,
    const float* __restrict__ addMat,
    const float* __restrict__ bnScale, const float* __restrict__ bnShift,
    int M, int N, int K)
{
  __shared__ __attribute__((aligned(16))) _Float16 As[BM][BK];    // (m,k) 4 KB
  __shared__ __attribute__((aligned(16))) _Float16 Bs[BNT][BK];   // (n,k) 4 KB

  int bz = blockIdx.z;
  const float* Ab = A + (long long)bz * aStride;
  const float* Bb = B + (long long)bz * bStride;
  float*       Cb = C + (long long)bz * cStride;

  int tid  = threadIdx.x;
  int m0   = blockIdx.y * BM;
  int n0   = blockIdx.x * BNT;
  int wave = tid >> 5;
  int lane = tid & 31;
  int wm   = wave >> 1;       // 0..3 -> 16-row tile
  int wn   = wave & 1;        // 0..1 -> 32-col strip (two 16-col subtiles)
  int half = lane >> 4;
  int l16  = lane & 15;

  v8f acc0 = {}, acc1 = {};

  // A staging map: 1 row x 8 consecutive k per thread
  int arow = tid >> 2;
  int acol = (tid & 3) * 8;
  int agm  = m0 + arow;
  int cagm = agm < M ? agm : (M - 1);
  const float* aptr = Ab + (long long)cagm * lda;

  for (int k0 = 0; k0 < K; k0 += BK) {
    // --- prefetch next K-chunk of B (uniform scalar branch, clamped addrs) ---
    if (k0 + BK < K) {
      if (BMODE == 2) {
        int n  = tid >> 2;
        int kc = (tid & 3) * 8;
        int gn  = n0 + n;
        int cgn = gn < N ? gn : (N - 1);
        int pk  = k0 + BK + kc;
        int cpk = pk < K ? pk : (K - 1);
        __builtin_prefetch(Bb + (long long)cgn * ldb + cpk, 0, 3);
      } else {
        int k   = tid >> 4;
        int n4  = (tid & 15) * 4;
        int pk  = k0 + BK + k;
        int cpk = pk < K ? pk : (K - 1);
        __builtin_prefetch(Bb + (long long)cpk * ldb + n0 + n4, 0, 3);
      }
    }
    // --- stage A tile (64x32): clamped unconditional loads, one b128 LDS store
    {
      v8h av;
#pragma unroll
      for (int j = 0; j < 8; ++j) {
        int gk  = k0 + acol + j;
        int cgk = gk < K ? gk : (K - 1);
        float v = aptr[cgk];
        bool ok = (agm < M) & (gk < K);
        av[j] = (_Float16)(ok ? v : 0.f);
      }
      *(v8h*)(&As[arow][acol]) = av;
    }
    // --- stage B tile (64x32) transposed into (n,k) ---
    if (BMODE == 2) {
      // B2 is (N x K) row-major: 8 consecutive k per thread at one n
      int n   = tid >> 2;
      int kc  = (tid & 3) * 8;
      int gn  = n0 + n;
      int cgn = gn < N ? gn : (N - 1);
      const float* bp = Bb + (long long)cgn * ldb;
      v8h bv;
#pragma unroll
      for (int j = 0; j < 8; ++j) {
        int gk  = k0 + kc + j;
        int cgk = gk < K ? gk : (K - 1);
        float v = bp[cgk];
        bool ok = (gn < N) & (gk < K);
        bv[j] = (_Float16)(ok ? v : 0.f);
      }
      *(v8h*)(&Bs[n][kc]) = bv;
    } else {
      // B is (K x N) row-major, ldb%4==0, N%64==0: float4 loads, coalesced
#pragma unroll
      for (int p = 0; p < 2; ++p) {
        int k   = (tid >> 4) + p * 16;
        int n4  = (tid & 15) * 4;
        int gk  = k0 + k;
        int cgk = gk < K ? gk : (K - 1);
        float4 vv = *(const float4*)(Bb + (long long)cgk * ldb + n0 + n4);
        float x0 = vv.x, x1 = vv.y, x2 = vv.z, x3 = vv.w;
        if (BMODE == 1) {
          float sc = bnScale[cgk], sh = bnShift[cgk];
          x0 = fmaxf(x0 * sc + sh, 0.f);
          x1 = fmaxf(x1 * sc + sh, 0.f);
          x2 = fmaxf(x2 * sc + sh, 0.f);
          x3 = fmaxf(x3 * sc + sh, 0.f);
        }
        bool ok = gk < K;
        Bs[n4 + 0][k] = (_Float16)(ok ? x0 : 0.f);
        Bs[n4 + 1][k] = (_Float16)(ok ? x1 : 0.f);
        Bs[n4 + 2][k] = (_Float16)(ok ? x2 : 0.f);
        Bs[n4 + 3][k] = (_Float16)(ok ? x3 : 0.f);
      }
    }
    __syncthreads();

    // --- fragments: contiguous k = 16*half + e in both A and B (same order) ---
    v16h af, bf0, bf1;
    int am = wm * 16 + l16;
#pragma unroll
    for (int e = 0; e < 16; ++e) {
      int kk = (half << 4) + e;  // one 16-half contiguous run -> 2x ds_load_b128
      af[e] = As[am][kk];
    }
    int bnr = wn * 32 + l16;
#pragma unroll
    for (int e = 0; e < 16; ++e) bf0[e] = Bs[bnr][(half << 4) + e];
#pragma unroll
    for (int e = 0; e < 16; ++e) bf1[e] = Bs[bnr + 16][(half << 4) + e];

    acc0 = __builtin_amdgcn_wmma_f32_16x16x32_f16(false, af, false, bf0,
                                                  (short)0, acc0, false, false);
    acc1 = __builtin_amdgcn_wmma_f32_16x16x32_f16(false, af, false, bf1,
                                                  (short)0, acc1, false, false);
    __syncthreads();
  }

  // --- epilogue: D lane map m = r + 8*half, n = lane&15 ---
  // bias/addMat gathered with clamped rows, unconditional (batched loads, one
  // wait), then guarded stores.  gn < N always holds for our launches (N%64==0).
#pragma unroll
  for (int s = 0; s < 2; ++s) {
    int gn = n0 + wn * 32 + s * 16 + l16;
    float addv[8], bv[8];
    if (HASADD) {
#pragma unroll
      for (int r = 0; r < 8; ++r) {
        int gm  = m0 + wm * 16 + r + 8 * half;
        int cgm = gm < M ? gm : (M - 1);
        addv[r] = addMat[(long long)cgm * ldc + gn];
      }
    }
    if (HASBIAS) {
#pragma unroll
      for (int r = 0; r < 8; ++r) {
        int gm  = m0 + wm * 16 + r + 8 * half;
        int cgm = gm < M ? gm : (M - 1);
        bv[r] = bias[cgm];
      }
    }
#pragma unroll
    for (int r = 0; r < 8; ++r) {
      int gm = m0 + wm * 16 + r + 8 * half;
      if (gm < M && gn < N) {
        float v = (s == 0) ? acc0[r] : acc1[r];
        if (HASBIAS) v += bv[r];
        if (HASADD)  v += addv[r];
        Cb[(long long)gm * ldc + gn] = v;
      }
    }
  }
}

// ---------------------------------------------------------------------------
// Elementwise BN+ReLU (bn3 -> sf/qf)
// ---------------------------------------------------------------------------
__global__ __launch_bounds__(256) void bn_relu_ew_kernel(
    const float* __restrict__ X, float* __restrict__ Y,
    const float* __restrict__ sc, const float* __restrict__ sh,
    int C, long long total)
{
  long long i = blockIdx.x * 256LL + threadIdx.x;
  if (i >= total) return;
  int c = (int)((i / NPTS) % C);
  float v = X[i] * sc[c] + sh[c];
  Y[i] = v > 0.f ? v : 0.f;
}

// ---------------------------------------------------------------------------
// LayerNorm over C per point, writing (n,c) row-major; flat buffer then
// reinterpreted as (192,4096) row-major == torch's .view(B,-1,N).
// ---------------------------------------------------------------------------
__global__ __launch_bounds__(256) void ln_transpose_kernel(
    const float* __restrict__ G, float* __restrict__ U,
    const float* __restrict__ g, const float* __restrict__ b, int C, int N)
{
  int n = blockIdx.x;
  long long base = (long long)blockIdx.y * C * N;
  const float* Gb = G + base;
  float*       Ub = U + base;
  int c = threadIdx.x;
  float v = 0.f;
  if (c < C) v = Gb[(long long)c * N + n];
  __shared__ float rs[256], rq[256];
  rs[c] = (c < C) ? v : 0.f;
  rq[c] = (c < C) ? v * v : 0.f;
  __syncthreads();
  for (int off = 128; off > 0; off >>= 1) {
    if (c < off) { rs[c] += rs[c + off]; rq[c] += rq[c + off]; }
    __syncthreads();
  }
  float m   = rs[0] / (float)C;
  float var = rq[0] / (float)C - m * m;
  float inv = rsqrtf(var + EPSN);
  if (c < C) Ub[(long long)n * C + c] = (v - m) * inv * g[c] + b[c];
}

// ---------------------------------------------------------------------------
// Row softmax in place (one block per row)
// ---------------------------------------------------------------------------
__global__ __launch_bounds__(256) void softmax_rows_kernel(float* __restrict__ S, int cols)
{
  float* row = S + (long long)blockIdx.x * cols;
  __shared__ float red[256];
  int tid = threadIdx.x;
  float mx = -3.0e38f;
  for (int j = tid; j < cols; j += 256) mx = fmaxf(mx, row[j]);
  red[tid] = mx; __syncthreads();
  for (int off = 128; off > 0; off >>= 1) {
    if (tid < off) red[tid] = fmaxf(red[tid], red[tid + off]);
    __syncthreads();
  }
  mx = red[0];
  __syncthreads();
  float s = 0.f;
  for (int j = tid; j < cols; j += 256) { float e = __expf(row[j] - mx); row[j] = e; s += e; }
  red[tid] = s; __syncthreads();
  for (int off = 128; off > 0; off >>= 1) {
    if (tid < off) red[tid] += red[tid + off];
    __syncthreads();
  }
  float inv = 1.f / red[0];
  for (int j = tid; j < cols; j += 256) row[j] *= inv;
}

// ---------------------------------------------------------------------------
// Prototype pooling
// ---------------------------------------------------------------------------
__global__ __launch_bounds__(256) void mask_counts_kernel(const int* __restrict__ sy,
                                                          float* __restrict__ cnt)
{
  float c0 = 0.f, c1 = 0.f, c2 = 0.f;
  for (int i = threadIdx.x; i < 2 * 5 * NPTS; i += 256) {
    int y = sy[i];
    if (y == 0) c0 += 1.f;
    else if (i < 5 * NPTS) c1 += 1.f;
    else c2 += 1.f;
  }
  __shared__ float r0[256], r1[256], r2[256];
  r0[threadIdx.x] = c0; r1[threadIdx.x] = c1; r2[threadIdx.x] = c2;
  __syncthreads();
  for (int off = 128; off > 0; off >>= 1) {
    if ((int)threadIdx.x < off) {
      r0[threadIdx.x] += r0[threadIdx.x + off];
      r1[threadIdx.x] += r1[threadIdx.x + off];
      r2[threadIdx.x] += r2[threadIdx.x + off];
    }
    __syncthreads();
  }
  if (threadIdx.x == 0) { cnt[0] = r0[0]; cnt[1] = r1[0]; cnt[2] = r2[0]; }
}

__global__ __launch_bounds__(256) void proto_sums_kernel(
    const float* __restrict__ sf, const int* __restrict__ sy,
    const float* __restrict__ cnt, float* __restrict__ fmRaw)
{
  int d = blockIdx.x;
  float s0 = 0.f, s1 = 0.f, s2 = 0.f;
  for (int i = threadIdx.x; i < 10 * NPTS; i += 256) {
    int bidx = i >> 12;
    int n    = i & (NPTS - 1);
    float v  = sf[((long long)bidx * 192 + d) * NPTS + n];
    int y = sy[i];
    if (y == 0) s0 += v;
    else if (bidx < 5) s1 += v;
    else s2 += v;
  }
  __shared__ float r0[256], r1[256], r2[256];
  r0[threadIdx.x] = s0; r1[threadIdx.x] = s1; r2[threadIdx.x] = s2;
  __syncthreads();
  for (int off = 128; off > 0; off >>= 1) {
    if ((int)threadIdx.x < off) {
      r0[threadIdx.x] += r0[threadIdx.x + off];
      r1[threadIdx.x] += r1[threadIdx.x + off];
      r2[threadIdx.x] += r2[threadIdx.x + off];
    }
    __syncthreads();
  }
  if (threadIdx.x == 0) {
    float cbg = cnt[0];
    float bg = r0[0] / fmaxf(cbg, 1.f);
    if (cbg < 1.f) bg = 0.1f;
    fmRaw[0 * 192 + d] = bg;
    fmRaw[1 * 192 + d] = r1[0] / fmaxf(cnt[1], 1.f);
    fmRaw[2 * 192 + d] = r2[0] / fmaxf(cnt[2], 1.f);
  }
}

__global__ __launch_bounds__(256) void fm_normalize_kernel(const float* __restrict__ raw,
                                                           float* __restrict__ fm)
{
  __shared__ float rs[256];
  for (int m = 0; m < 3; ++m) {
    float x = 0.f;
    if (threadIdx.x < 192) x = raw[m * 192 + threadIdx.x];
    rs[threadIdx.x] = x * x;
    __syncthreads();
    for (int off = 128; off > 0; off >>= 1) {
      if ((int)threadIdx.x < off) rs[threadIdx.x] += rs[threadIdx.x + off];
      __syncthreads();
    }
    float norm = sqrtf(rs[0]);
    if (threadIdx.x < 192) fm[m * 192 + threadIdx.x] = x / norm;
    __syncthreads();
  }
}

// ---------------------------------------------------------------------------
// est (2,8192,192): [mean-over-shot of corrected support]^T ++ [corrected query]^T
// ---------------------------------------------------------------------------
__global__ __launch_bounds__(256) void build_est_kernel(const float* __restrict__ seq,
                                                        float* __restrict__ est)
{
  long long idx = blockIdx.x * 256LL + threadIdx.x;
  if (idx >= 2LL * 8192 * 192) return;
  int d  = (int)(idx % 192);
  int n2 = (int)((idx / 192) % 8192);
  int b  = (int)(idx / (192LL * 8192));
  float v;
  if (n2 < NPTS) {
    float s = 0.f;
    for (int k = 0; k < 5; ++k)
      s += seq[((long long)(b * 5 + k) * 192 + d) * NPTS + n2];
    v = 0.2f * s;
  } else {
    v = seq[((long long)(10 + b) * 192 + d) * NPTS + (n2 - NPTS)];
  }
  est[idx] = v;
}

__global__ __launch_bounds__(256) void enhance_logits_kernel(
    const float* __restrict__ fm, const float* __restrict__ est, float* __restrict__ L)
{
  int idx = blockIdx.x * 256 + threadIdx.x;
  if (idx >= 2 * 3 * 8192) return;
  int n2 = idx % 8192;
  int m  = (idx / 8192) % 3;
  int b  = idx / (3 * 8192);
  const float* e = est + ((long long)b * 8192 + n2) * 192;
  const float* f = fm + m * 192;
  float acc = 0.f;
  for (int d = 0; d < 192; ++d) acc += f[d] * e[d];
  L[idx] = acc * 0.07216878364870323f;  // 1/sqrt(192)
}

__global__ __launch_bounds__(192) void fm2_kernel(
    const float* __restrict__ fm, const float* __restrict__ est,
    const float* __restrict__ L, float* __restrict__ fm2)
{
  int m = blockIdx.x, b = blockIdx.y, d = threadIdx.x;
  const float* a = L + ((long long)b * 3 + m) * 8192;
  const float* e = est + (long long)b * 8192 * 192 + d;
  float acc = 0.f;
  for (int n2 = 0; n2 < 8192; ++n2) acc += a[n2] * e[(long long)n2 * 192];
  fm2[((long long)b * 3 + m) * 192 + d] = fm[m * 192 + d] + acc;
}

// logits[b,n,m] = sum_d qf[b,d,n] * F[b*Fstride + m*192 + d]
__global__ __launch_bounds__(256) void logits_kernel(
    const float* __restrict__ qf, const float* __restrict__ F,
    long long Fstride, float* __restrict__ out)
{
  int i = blockIdx.x * 256 + threadIdx.x;
  if (i >= 2 * NPTS) return;
  int b = i >> 12;
  int n = i & (NPTS - 1);
  const float* Fb = F + (long long)b * Fstride;
  const float* q  = qf + (long long)b * 192 * NPTS + n;
  float a0 = 0.f, a1 = 0.f, a2 = 0.f;
  for (int d = 0; d < 192; ++d) {
    float v = q[(long long)d * NPTS];
    a0 += v * Fb[d];
    a1 += v * Fb[192 + d];
    a2 += v * Fb[384 + d];
  }
  out[(long long)i * 3 + 0] = a0;
  out[(long long)i * 3 + 1] = a1;
  out[(long long)i * 3 + 2] = a2;
}

// ---------------------------------------------------------------------------
// Orchestration
// ---------------------------------------------------------------------------
extern "C" void kernel_launch(void* const* d_in, const int* in_sizes, int n_in,
                              void* d_out, int out_size, void* d_ws, size_t ws_size,
                              hipStream_t stream)
{
  const float* supf    = (const float*)d_in[0];
  const float* supe    = (const float*)d_in[1];
  const int*   supy    = (const int*)  d_in[2];
  const float* qryf    = (const float*)d_in[3];
  const float* qrye    = (const float*)d_in[4];
  const float* bn1_g   = (const float*)d_in[6];
  const float* bn1_b   = (const float*)d_in[7];
  const float* conv1_W = (const float*)d_in[8];
  const float* conv1_b = (const float*)d_in[9];
  const float* bn2_g   = (const float*)d_in[10];
  const float* bn2_b   = (const float*)d_in[11];
  const float* conv2_W = (const float*)d_in[12];
  const float* conv2_b = (const float*)d_in[13];
  const float* bn3_g   = (const float*)d_in[14];
  const float* bn3_b   = (const float*)d_in[15];
  const float* pre1_W  = (const float*)d_in[16];
  const float* pre1_b  = (const float*)d_in[17];
  const float* ln1_g   = (const float*)d_in[18];
  const float* ln1_b   = (const float*)d_in[19];
  const float* pre2_W  = (const float*)d_in[20];
  const float* pre2_b  = (const float*)d_in[21];
  const float* ln2_g   = (const float*)d_in[22];
  const float* ln2_b   = (const float*)d_in[23];
  const float* dq_W    = (const float*)d_in[24];
  const float* dq_b    = (const float*)d_in[25];
  const float* dv_W    = (const float*)d_in[26];
  const float* dv_b    = (const float*)d_in[27];
  float* out = (float*)d_out;

  const int N = NPTS;
  float* ws = (float*)d_ws;
  size_t off = 0;
  auto alloc = [&](size_t n) { float* p = ws + off; off += n; return p; };
  // ~177 MB total workspace (with reuse below)
  float* Y1  = alloc((size_t)10 * 196 * N);   // conv1 out / pre-GEMM out / est (later)
  float* Y2  = alloc((size_t)10 * 192 * N);   // conv2 out / Q,V,S,Lbuf (later)
  float* sf  = alloc((size_t)10 * 192 * N);
  float* qf  = alloc((size_t)2  * 192 * N);
  float* Es2 = alloc((size_t)12 * 192 * N);
  float* SEQ = alloc((size_t)12 * 192 * N);
  float* sc1 = alloc(320); float* sh1 = alloc(320);
  float* sc2 = alloc(256); float* sh2 = alloc(256);
  float* sc3 = alloc(256); float* sh3 = alloc(256);
  float* cnt = alloc(16);
  float* fmR = alloc(576);
  float* fm  = alloc(576);
  float* fm2 = alloc(1152);
  // region reuse (sequentially disjoint lifetimes)
  float* Qb  = Y2;
  float* Vb  = Y2 + (size_t)192 * N;
  float* Sb  = Vb + (size_t)192 * N;
  float* Lb  = Sb + 192 * 192;
  float* est = Y1;

  // ---- support bn_fc ----
  bn_stats_kernel<<<320, 256, 0, stream>>>(supf, 10, 320, N, bn1_g, bn1_b, sc1, sh1);
  gemm_wmma<1, true, false><<<dim3(N / BNT, 4, 10), 256, 0, stream>>>(
      conv1_W, 0, 320, supf, (long long)320 * N, N, Y1, (long long)196 * N, N,
      conv1_b, nullptr, sc1, sh1, 196, N, 320);
  bn_stats_kernel<<<196, 256, 0, stream>>>(Y1, 10, 196, N, bn2_g, bn2_b, sc2, sh2);
  gemm_wmma<1, true, false><<<dim3(N / BNT, 3, 10), 256, 0, stream>>>(
      conv2_W, 0, 196, Y1, (long long)196 * N, N, Y2, (long long)192 * N, N,
      conv2_b, nullptr, sc2, sh2, 192, N, 196);
  bn_stats_kernel<<<192, 256, 0, stream>>>(Y2, 10, 192, N, bn3_g, bn3_b, sc3, sh3);
  {
    long long tot = (long long)10 * 192 * N;
    bn_relu_ew_kernel<<<(unsigned)((tot + 255) / 256), 256, 0, stream>>>(Y2, sf, sc3, sh3, 192, tot);
  }

  // ---- query bn_fc (separate batch stats) ----
  bn_stats_kernel<<<320, 256, 0, stream>>>(qryf, 2, 320, N, bn1_g, bn1_b, sc1, sh1);
  gemm_wmma<1, true, false><<<dim3(N / BNT, 4, 2), 256, 0, stream>>>(
      conv1_W, 0, 320, qryf, (long long)320 * N, N, Y1, (long long)196 * N, N,
      conv1_b, nullptr, sc1, sh1, 196, N, 320);
  bn_stats_kernel<<<196, 256, 0, stream>>>(Y1, 2, 196, N, bn2_g, bn2_b, sc2, sh2);
  gemm_wmma<1, true, false><<<dim3(N / BNT, 3, 2), 256, 0, stream>>>(
      conv2_W, 0, 196, Y1, (long long)196 * N, N, Y2, (long long)192 * N, N,
      conv2_b, nullptr, sc2, sh2, 192, N, 196);
  bn_stats_kernel<<<192, 256, 0, stream>>>(Y2, 2, 192, N, bn3_g, bn3_b, sc3, sh3);
  {
    long long tot = (long long)2 * 192 * N;
    bn_relu_ew_kernel<<<(unsigned)((tot + 255) / 256), 256, 0, stream>>>(Y2, qf, sc3, sh3, 192, tot);
  }

  // ---- prototype pooling -> fm (3,192), L2-normalized rows ----
  mask_counts_kernel<<<1, 256, 0, stream>>>(supy, cnt);
  proto_sums_kernel<<<192, 256, 0, stream>>>(sf, supy, cnt, fmR);
  fm_normalize_kernel<<<1, 256, 0, stream>>>(fmR, fm);

  // ---- pre (GEMM + LN, store (n,c) == reinterpreted (192,4096)) ----
  gemm_wmma<0, true, false><<<dim3(N / BNT, 3, 10), 256, 0, stream>>>(
      pre1_W, 0, 117, supe, (long long)117 * N, N, Y1, (long long)192 * N, N,
      pre1_b, nullptr, nullptr, nullptr, 192, N, 117);
  ln_transpose_kernel<<<dim3(N, 10), 256, 0, stream>>>(Y1, Es2, ln1_g, ln1_b, 192, N);
  gemm_wmma<0, true, false><<<dim3(N / BNT, 3, 2), 256, 0, stream>>>(
      pre2_W, 0, 117, qrye, (long long)117 * N, N, Y1, (long long)192 * N, N,
      pre2_b, nullptr, nullptr, nullptr, 192, N, 117);
  ln_transpose_kernel<<<dim3(N, 2), 256, 0, stream>>>(Y1, Es2 + (size_t)10 * 192 * N,
                                                      ln2_g, ln2_b, 192, N);

  // ---- correct(): per-batch channel attention ----
  for (int bi = 0; bi < 12; ++bi) {
    const float* E = Es2 + (size_t)bi * 192 * N;
    gemm_wmma<0, true, false><<<dim3(N / BNT, 3, 1), 256, 0, stream>>>(
        dq_W, 0, 192, E, 0, N, Qb, 0, N, dq_b, nullptr, nullptr, nullptr, 192, N, 192);
    gemm_wmma<0, true, false><<<dim3(N / BNT, 3, 1), 256, 0, stream>>>(
        dv_W, 0, 192, E, 0, N, Vb, 0, N, dv_b, nullptr, nullptr, nullptr, 192, N, 192);
    gemm_wmma<2, false, false><<<dim3(192 / BNT, 3, 1), 256, 0, stream>>>(
        Qb, 0, N, E, 0, N, Sb, 0, 192, nullptr, nullptr, nullptr, nullptr, 192, 192, N);
    softmax_rows_kernel<<<192, 256, 0, stream>>>(Sb, 192);
    gemm_wmma<0, false, true><<<dim3(N / BNT, 3, 1), 256, 0, stream>>>(
        Sb, 0, 192, Vb, 0, N, SEQ + (size_t)bi * 192 * N, 0, N,
        nullptr, E, nullptr, nullptr, 192, N, 192);
  }

  // ---- enhance: prototypes cross-attend to est ----
  {
    long long tot = 2LL * 8192 * 192;
    build_est_kernel<<<(unsigned)((tot + 255) / 256), 256, 0, stream>>>(SEQ, est);
  }
  enhance_logits_kernel<<<(2 * 3 * 8192 + 255) / 256, 256, 0, stream>>>(fm, est, Lb);
  softmax_rows_kernel<<<6, 256, 0, stream>>>(Lb, 8192);
  fm2_kernel<<<dim3(3, 2), 192, 0, stream>>>(fm, est, Lb, fm2);

  // ---- logits ----
  logits_kernel<<<(2 * NPTS + 255) / 256, 256, 0, stream>>>(qf, fm, 0, out);
  logits_kernel<<<(2 * NPTS + 255) / 256, 256, 0, stream>>>(qf, fm2, 576, out + (size_t)2 * NPTS * 3);
}
